// AttentionLayer_28123445854653
// MI455X (gfx1250) — matrix operand order repaired and checked by reading, not compiled, not run on hardware
//
#include <hip/hip_runtime.h>

typedef __attribute__((ext_vector_type(16))) __bf16 v16bf;
typedef __attribute__((ext_vector_type(8)))  float  v8f;
typedef unsigned int u32;

#define DIM  256
#define NTOK 4096

union Frag { uint4 q[2]; v16bf v; };

__device__ __forceinline__ unsigned short f2bf(float f) {
  u32 u = __float_as_uint(f);
  u32 r = (u + 0x7FFFu + ((u >> 16) & 1u)) >> 16;
  return (unsigned short)r;
}
__device__ __forceinline__ u32 packbf(float a, float b) {
  return (u32)f2bf(a) | ((u32)f2bf(b) << 16);
}

// Async global -> LDS copy (16 bytes), GVS mode: uniform 64-bit SGPR base +
// per-lane 32-bit byte offset. Tracked by ASYNCcnt. No VGPR round-trip.
__device__ __forceinline__ void async_cp_b128(u32 lds_byte_addr, u32 goff,
                                              const void* base) {
  asm volatile("global_load_async_to_lds_b128 %0, %1, %2"
               :
               : "v"(lds_byte_addr), "v"(goff), "s"(base)
               : "memory");
}
__device__ __forceinline__ void wait_async0() {
  asm volatile("s_wait_asynccnt 0x0" ::: "memory");
}

// ---------------------------------------------------------------------------
// Kernel 1: projection  out[n,d] = sum_k xt[n,k] * W[k,d] + bias[d]
// Each block: one batch, 64 token rows, full 256 cols. 256 threads = 8 waves.
// Output written as pre-swizzled bf16 WMMA fragments:
//   mode 0 (Q): A-fragment layout   Qs[b][nt][kk][lane][v]
//   mode 1 (K): B-fragment layout   Ks[b][mt][kk][lane][v]   (K-dim = d)
//   mode 2 (V): B-fragment layout   Vs[b][kv][dt][lane][v]   (K-dim = m)
// ---------------------------------------------------------------------------
__global__ __launch_bounds__(256) void proj_kernel(
    const float* __restrict__ x, const float* __restrict__ W,
    const float* __restrict__ bias, u32* __restrict__ outswz, int mode) {
  extern __shared__ u32 lds[];
  u32* aLds  = lds;                 // 64 rows x 132 dwords (padded)   = 8448
  u32* wfrag = lds + 8448;          // 16 tiles x 256 dwords           = 4096
  u32* outDw = lds + 8448;          // reused after compute: 64x128    = 8192
  unsigned short* outH = (unsigned short*)outDw;

  const int tid = threadIdx.x, lane = tid & 31, wv = tid >> 5;
  const int lh = lane >> 4, l15 = lane & 15;
  const int rb = wv >> 1, ch = wv & 1;
  const int b = blockIdx.y;
  const int n0 = blockIdx.x * 64;

  // Stage x tile (transposed on the fly, f32 -> bf16 pairs along d)
  for (int i = 0; i < 32; ++i) {
    int f = i * 256 + tid;
    int nl = f & 63, dj = f >> 6;                       // dj = d/2
    float x0 = x[((size_t)b * DIM + 2 * dj)     * NTOK + n0 + nl];
    float x1 = x[((size_t)b * DIM + 2 * dj + 1) * NTOK + n0 + nl];
    aLds[nl * 132 + dj] = packbf(x0, x1);
  }

  v8f zero = {};
  v8f acc[8];
#pragma unroll
  for (int j = 0; j < 8; ++j) acc[j] = zero;

  for (int kk = 0; kk < 8; ++kk) {
    __syncthreads();
    // Stage W[32k : 32k+32, :] in B-fragment layout
    for (int i = 0; i < 16; ++i) {
      int f = i * 256 + tid;
      int t = f >> 8, r = f & 255, ls = r >> 3, v = r & 7;
      int k = 32 * kk + 2 * (v & 3) + 8 * (ls >> 4) + 16 * (v >> 2);
      int n = 16 * t + (ls & 15);
      wfrag[f] = packbf(W[(size_t)k * DIM + n], W[(size_t)(k + 1) * DIM + n]);
    }
    __syncthreads();

    // A fragment (16x32 bf16) from LDS
    int arow = 16 * rb + l15;
    const u32* ap = aLds + arow * 132 + 16 * kk + lh * 4;
    Frag fa;
    fa.q[0] = *(const uint4*)ap;
    fa.q[1] = *(const uint4*)(ap + 8);

#pragma unroll
    for (int j = 0; j < 8; ++j) {
      v16bf bfr = *(const v16bf*)(wfrag + (8 * ch + j) * 256 + lane * 8);
      acc[j] = __builtin_amdgcn_wmma_f32_16x16x32_bf16(
          false, fa.v, false, bfr, (short)0, acc[j], false, false);
    }
  }

  __syncthreads();
  // C fragments + bias -> bf16 tile in LDS (row = token, col = d)
#pragma unroll
  for (int j = 0; j < 8; ++j) {
    int col = 128 * ch + 16 * j + l15;
    float bb = bias[col];
#pragma unroll
    for (int i = 0; i < 8; ++i) {
      int row = 16 * rb + i + 8 * lh;
      outH[row * 256 + col] = f2bf(acc[j][i] + bb);
    }
  }
  __syncthreads();

  // Swizzle-store fragments to global (coalesced dwords)
  if (mode == 0) {            // Q: A-fragment layout
    for (int i = 0; i < 32; ++i) {
      int f = i * 256 + tid;
      int v = f & 7, ls = (f >> 3) & 31, kk = (f >> 8) & 7, ntl = f >> 11;
      int lsh = ls >> 4;
      int row = 16 * ntl + (ls & 15);
      int dw  = 16 * kk + (v & 3) + (v >> 2) * 8 + lsh * 4;
      size_t gi = ((((size_t)b * 256 + (4 * blockIdx.x + ntl)) * 8 + kk) * 256)
                  + ls * 8 + v;
      outswz[gi] = outDw[row * 128 + dw];
    }
  } else if (mode == 1) {     // K: B-fragment layout (pairs along d)
    for (int i = 0; i < 32; ++i) {
      int f = i * 256 + tid;
      int v = f & 7, ls = (f >> 3) & 31, kk = (f >> 8) & 7, mtl = f >> 11;
      int lsh = ls >> 4;
      int row = 16 * mtl + (ls & 15);
      int dw  = 16 * kk + (v & 3) + (v >> 2) * 8 + lsh * 4;
      size_t gi = ((((size_t)b * 256 + (4 * blockIdx.x + mtl)) * 8 + kk) * 256)
                  + ls * 8 + v;
      outswz[gi] = outDw[row * 128 + dw];
    }
  } else {                    // V: B-fragment layout (pairs along token m)
    for (int i = 0; i < 32; ++i) {
      int f = i * 256 + tid;
      int v = f & 7, ls = (f >> 3) & 31, dt = (f >> 8) & 15, kl = f >> 12;
      int lsh = ls >> 4;
      int m = 32 * kl + 2 * (v & 3) + 8 * lsh + 16 * (v >> 2);
      int d = 16 * dt + (ls & 15);
      u32 val = (u32)outH[m * 256 + d] | ((u32)outH[(m + 1) * 256 + d] << 16);
      size_t gi = ((((size_t)b * 128 + (2 * blockIdx.x + kl)) * 16 + dt) * 256)
                  + ls * 8 + v;
      outswz[gi] = val;
    }
  }
}

// ---------------------------------------------------------------------------
// Kernel 2: flash attention. Block = 128 threads (4 waves), 64 query rows
// (16 per wave, kept as 8 A-fragments in registers). Key blocks of 64 staged
// to double-buffered LDS with GLOBAL_LOAD_ASYNC_TO_LDS_B128 (overlaps the
// WMMA stream on the other buffer). Online softmax; O accumulated in f32.
// LDS: buf0 K+V 64KB | buf1 K+V 64KB | per-wave P scratch 8KB  = 136KB.
// ---------------------------------------------------------------------------
__global__ __launch_bounds__(128) void attn_kernel(
    const u32* __restrict__ Qs, const u32* __restrict__ Ks,
    const u32* __restrict__ Vs, float* __restrict__ out) {
  extern __shared__ u32 lds[];
  const int tid = threadIdx.x, lane = tid & 31, wv = tid >> 5;
  const int lh = lane >> 4, l15 = lane & 15;
  const int b = blockIdx.y;
  const int n0 = blockIdx.x * 64;
  const int nt = blockIdx.x * 4 + wv;

  u32* myP = lds + 32768 + wv * 512;      // per-wave P scratch: 16 x 32 dw
  unsigned short* myPH = (unsigned short*)myP;

  // Issue async staging of key-block jb into LDS buffer `buf` (no wait).
  auto stage = [&](int jb, int buf) {
    const u32* gk = Ks + (((size_t)b * 256 + 4 * jb) * 8) * 256;
    const u32* gv = Vs + (((size_t)b * 128 + 2 * jb) * 16) * 256;
    u32 lk = (u32)(uintptr_t)(lds + buf * 16384);
    u32 lv = lk + 32768;                  // +8192 dwords
#pragma unroll
    for (int i = 0; i < 16; ++i) {
      u32 off = (u32)(i * 128 + tid) * 16u;
      async_cp_b128(lk + off, off, gk);
      async_cp_b128(lv + off, off, gv);
    }
  };

  // Q fragments for this wave's 16 rows (full D=256)
  v16bf qa[8];
#pragma unroll
  for (int kk = 0; kk < 8; ++kk)
    qa[kk] = *(const v16bf*)(Qs + ((((size_t)b * 256 + nt) * 8 + kk) * 256)
                             + lane * 8);

  v8f zero = {};
  v8f o[16];
#pragma unroll
  for (int dt = 0; dt < 16; ++dt) o[dt] = zero;
  float mrun[8], lrun[8];
#pragma unroll
  for (int i = 0; i < 8; ++i) { mrun[i] = -1e30f; lrun[i] = 0.0f; }

  // Prologue: fill buffer 0
  stage(0, 0);
  wait_async0();
  __syncthreads();

  for (int jb = 0; jb < 64; ++jb) {
    const int cur = jb & 1;
    if (jb + 1 < 64) stage(jb + 1, cur ^ 1);   // background DMA into other buf

    u32* ldsKb = lds + cur * 16384;
    u32* ldsVb = ldsKb + 8192;

    // S = Q K^T : 4 key tiles x 8 k-chunks
    v8f s[4];
#pragma unroll
    for (int mt = 0; mt < 4; ++mt) s[mt] = zero;
#pragma unroll
    for (int kk = 0; kk < 8; ++kk) {
#pragma unroll
      for (int mt = 0; mt < 4; ++mt) {
        v16bf kb = *(const v16bf*)(ldsKb + (mt * 8 + kk) * 256 + lane * 8);
        s[mt] = __builtin_amdgcn_wmma_f32_16x16x32_bf16(
            false, qa[kk], false, kb, (short)0, s[mt], false, false);
      }
    }

    // Online softmax: rows live on 16-lane halves of the wave
#pragma unroll
    for (int i = 0; i < 8; ++i) {
      float mx = fmaxf(fmaxf(s[0][i], s[1][i]), fmaxf(s[2][i], s[3][i]));
      mx = fmaxf(mx, __shfl_xor(mx, 1, 32));
      mx = fmaxf(mx, __shfl_xor(mx, 2, 32));
      mx = fmaxf(mx, __shfl_xor(mx, 4, 32));
      mx = fmaxf(mx, __shfl_xor(mx, 8, 32));
      float mnew = fmaxf(mrun[i], mx);
      float c = __expf(mrun[i] - mnew);
      mrun[i] = mnew;
      float sum = 0.0f;
#pragma unroll
      for (int mt = 0; mt < 4; ++mt) {
        float p = __expf(s[mt][i] - mnew);
        s[mt][i] = p;
        sum += p;
      }
      sum += __shfl_xor(sum, 1, 32);
      sum += __shfl_xor(sum, 2, 32);
      sum += __shfl_xor(sum, 4, 32);
      sum += __shfl_xor(sum, 8, 32);
      lrun[i] = lrun[i] * c + sum;
#pragma unroll
      for (int dt = 0; dt < 16; ++dt) o[dt][i] *= c;
    }

    // P (C-fragment layout) -> bf16 A-fragment via wave-private LDS
#pragma unroll
    for (int mt = 0; mt < 4; ++mt)
#pragma unroll
      for (int i = 0; i < 8; ++i)
        myPH[(i + 8 * lh) * 64 + 16 * mt + l15] = f2bf(s[mt][i]);

    // O += P @ V
#pragma unroll
    for (int kv = 0; kv < 2; ++kv) {
      Frag pf;
      const u32* pp = myP + l15 * 32 + 16 * kv + lh * 4;
      pf.q[0] = *(const uint4*)pp;
      pf.q[1] = *(const uint4*)(pp + 8);
#pragma unroll
      for (int dt = 0; dt < 16; ++dt) {
        v16bf vb = *(const v16bf*)(ldsVb + (kv * 16 + dt) * 256 + lane * 8);
        o[dt] = __builtin_amdgcn_wmma_f32_16x16x32_bf16(
            false, pf.v, false, vb, (short)0, o[dt], false, false);
      }
    }

    // Drain this wave's async copies, then make buffer swap visible.
    wait_async0();
    __syncthreads();
  }

  // Normalize, transpose to (d, n) through LDS, store coalesced
  float rl[8];
#pragma unroll
  for (int i = 0; i < 8; ++i) rl[i] = 1.0f / lrun[i];
  float* outT = (float*)lds;              // 256 d-rows x 64 tokens
#pragma unroll
  for (int dt = 0; dt < 16; ++dt) {
    int d = 16 * dt + l15;
#pragma unroll
    for (int i = 0; i < 8; ++i) {
      int nl = 16 * wv + i + 8 * lh;
      outT[d * 64 + nl] = o[dt][i] * rl[i];
    }
  }
  __syncthreads();
#pragma unroll
  for (int it = 0; it < 32; ++it) {
    int idx4 = it * 128 + tid;
    int d = idx4 >> 4, c4 = idx4 & 15;
    float4 val = ((const float4*)outT)[idx4];
    *(float4*)(out + ((size_t)b * 256 + d) * 4096 + n0 + c4 * 4) = val;
  }
}

// ---------------------------------------------------------------------------
extern "C" void kernel_launch(void* const* d_in, const int* in_sizes, int n_in,
                              void* d_out, int out_size, void* d_ws, size_t ws_size,
                              hipStream_t stream) {
  (void)in_sizes; (void)n_in; (void)out_size; (void)ws_size;
  const float* x  = (const float*)d_in[0];
  const float* Wq = (const float*)d_in[1];
  const float* bq = (const float*)d_in[2];
  const float* Wk = (const float*)d_in[3];
  const float* bk = (const float*)d_in[4];
  const float* Wv = (const float*)d_in[5];
  const float* bv = (const float*)d_in[6];
  float* out = (float*)d_out;

  // Workspace: 3 x (B * N * D) bf16 = 3 x 8 MB, swizzled fragment layout
  u32* Qs = (u32*)d_ws;
  u32* Ks = Qs + (size_t)4 * 256 * 8 * 256;
  u32* Vs = Ks + (size_t)4 * 256 * 8 * 256;

  dim3 g1(64, 4);
  proj_kernel<<<g1, 256, 66560, stream>>>(x, Wq, bq, Qs, 0);
  proj_kernel<<<g1, 256, 66560, stream>>>(x, Wk, bk, Ks, 1);
  proj_kernel<<<g1, 256, 66560, stream>>>(x, Wv, bv, Vs, 2);

  dim3 g2(64, 4);
  attn_kernel<<<g2, 128, 139264, stream>>>(Qs, Ks, Vs, out);
}
